// DiffConvAdaptive_55070070669778
// MI455X (gfx1250) — compile-verified, matrix operand order checked by reading, not run on hardware
//
#include <hip/hip_runtime.h>
#include <hip/hip_bf16.h>

// ---------------------------------------------------------------------------
// DiffConvAdaptive for MI455X (gfx1250, wave32, WMMA + async-to-LDS)
//
//   0) fp32 -> bf16 weight conversion (p1_w, p2_w, proj_w)
//   1) adaptive-pool x over N into 9 segments (pool commutes with p1 GEMM)
//   2) tiny WMMA GEMM: pooled(288x768) x p1_w^T -> g (fp32)       [guarded]
//   3) big  WMMA GEMM: x(32768x768)   x p2_w^T -> x1 (bf16)       [cvt-on-load]
//   4) kernel-gen: SiLU(g) x kg_w^T + kg_b, residual-mean subtraction
//   5) depthwise 3x3 conv on x1 with per-(b,c) kernels -> y (bf16)
//   6) big  WMMA GEMM: y x proj_w^T -> d_out (fp32)               [ASYNC staging]
//
// GEMM: bf16 operands / fp32 accumulation (v_wmma_f32_16x16x32_bf16),
// block tile 128x128, K-step 32, double-buffered LDS. When A is already bf16
// and tiles are full, tiles are staged with global_load_async_to_lds_b128
// (ASYNCcnt) so the tile DMA overlaps the WMMA stream of the previous tile.
// ---------------------------------------------------------------------------

typedef __attribute__((ext_vector_type(16))) __bf16 v16bf;
typedef __attribute__((ext_vector_type(8)))  float  v8f;
typedef __attribute__((ext_vector_type(4)))  float  f32x4;

union FragU { v16bf v; f32x4 f[2]; };
union Pack16 { __bf16 h[16]; f32x4 f[2]; };

// out[m, n] = sum_k A[m, k] * W[n, k] + bias[n]
// A: fp32 (CVT_A) or bf16; W: bf16, k-contiguous. K mult of 32, N mult of 128.
// FULL: M is a multiple of 128 (no row guards). ASYNC staging when bf16+FULL.
template <bool CVT_A, bool OUT_BF16, bool FULL>
__global__ __launch_bounds__(256) void gemm_wmma_bf16(
    const void* __restrict__ Av, const __bf16* __restrict__ W,
    const float* __restrict__ bias, void* __restrict__ outv,
    int M, int K, int N)
{
    constexpr bool ASYNC = (!CVT_A) && FULL;

    __shared__ __align__(16) __bf16 sA[2][128 * 32];
    __shared__ __align__(16) __bf16 sW[2][128 * 32];

    const int tid    = threadIdx.x;
    const int lane   = tid & 31;
    const int waveId = tid >> 5;          // 0..7
    const int waveM  = waveId >> 2;       // 0..1  (64-row slab)
    const int waveN  = waveId & 3;        // 0..3  (32-col slab)

    // staging role: each thread moves 16 bf16 (half of one tile row)
    const int ldRow = tid >> 1;           // 0..127
    const int ldK   = (tid & 1) << 4;     // 0 or 16

    const long gRowA = (long)blockIdx.y * 128 + ldRow;
    const long gRowW = (long)blockIdx.x * 128 + ldRow;
    const bool aInRange = FULL ? true : (gRowA < M);

    // WMMA fragment addressing (ISA 16-bit A-matrix layout, wave32):
    // lanes 0-15 hold K {0..7,16..23}, lanes 16-31 hold K {8..15,24..31}
    const int khalf = (lane & 16) ? 8 : 0;
    const int mrow  = lane & 15;
    const int aBase = waveM * 64;
    const int bBase = waveN * 32;

    v8f acc[4][2] = {};

    // register staging (regular path)
    f32x4 ra0 = {0.f, 0.f, 0.f, 0.f}, ra1 = {0.f, 0.f, 0.f, 0.f};
    f32x4 rw0 = {0.f, 0.f, 0.f, 0.f}, rw1 = {0.f, 0.f, 0.f, 0.f};

    auto loadRegs = [&](int k0) {
        if (aInRange) {
            if constexpr (CVT_A) {
                const f32x4* p = (const f32x4*)((const float*)Av +
                                   gRowA * (long)K + k0 + ldK);
                f32x4 f0 = p[0], f1 = p[1], f2 = p[2], f3 = p[3];
                Pack16 pk;
                #pragma unroll
                for (int e = 0; e < 4; ++e) {
                    pk.h[e]      = (__bf16)f0[e];
                    pk.h[4 + e]  = (__bf16)f1[e];
                    pk.h[8 + e]  = (__bf16)f2[e];
                    pk.h[12 + e] = (__bf16)f3[e];
                }
                ra0 = pk.f[0]; ra1 = pk.f[1];
            } else {
                const f32x4* p = (const f32x4*)((const __bf16*)Av +
                                   gRowA * (long)K + k0 + ldK);
                ra0 = p[0]; ra1 = p[1];
            }
        } else {
            ra0 = f32x4{0.f, 0.f, 0.f, 0.f};
            ra1 = f32x4{0.f, 0.f, 0.f, 0.f};
        }
        {
            const f32x4* p = (const f32x4*)(W + gRowW * (long)K + k0 + ldK);
            rw0 = p[0]; rw1 = p[1];
        }
    };

    auto storeRegs = [&](int bi) {
        f32x4* da = (f32x4*)(&sA[bi][ldRow * 32 + ldK]);
        da[0] = ra0; da[1] = ra1;
        f32x4* dw = (f32x4*)(&sW[bi][ldRow * 32 + ldK]);
        dw[0] = rw0; dw[1] = rw1;
    };

    auto stageAsync = [&](int bi, int k0) {
        // per-lane async DMA: 2 x b128 for A-tile slot, 2 x b128 for W-tile slot
        unsigned ldsA = (unsigned)(unsigned long long)(void*)&sA[bi][ldRow * 32 + ldK];
        unsigned ldsW = (unsigned)(unsigned long long)(void*)&sW[bi][ldRow * 32 + ldK];
        unsigned long long ga =
            (unsigned long long)((const __bf16*)Av + gRowA * (long)K + k0 + ldK);
        unsigned long long gw =
            (unsigned long long)(W + gRowW * (long)K + k0 + ldK);
        asm volatile("global_load_async_to_lds_b128 %0, %1, off"
                     :: "v"(ldsA), "v"(ga) : "memory");
        asm volatile("global_load_async_to_lds_b128 %0, %1, off offset:16"
                     :: "v"(ldsA), "v"(ga) : "memory");
        asm volatile("global_load_async_to_lds_b128 %0, %1, off"
                     :: "v"(ldsW), "v"(gw) : "memory");
        asm volatile("global_load_async_to_lds_b128 %0, %1, off offset:16"
                     :: "v"(ldsW), "v"(gw) : "memory");
    };

    auto compute = [&](int bi) {
        const __bf16* bA = sA[bi];
        const __bf16* bW = sW[bi];
        FragU fa[4], fb[2];
        #pragma unroll
        for (int mi = 0; mi < 4; ++mi) {
            const f32x4* p = (const f32x4*)(bA + (aBase + mi * 16 + mrow) * 32 + khalf);
            fa[mi].f[0] = p[0];          // K khalf .. khalf+7
            fa[mi].f[1] = p[2];          // K khalf+16 .. khalf+23
        }
        #pragma unroll
        for (int ni = 0; ni < 2; ++ni) {
            const f32x4* p = (const f32x4*)(bW + (bBase + ni * 16 + mrow) * 32 + khalf);
            fb[ni].f[0] = p[0];
            fb[ni].f[1] = p[2];
        }
        #pragma unroll
        for (int mi = 0; mi < 4; ++mi)
            #pragma unroll
            for (int ni = 0; ni < 2; ++ni)
                acc[mi][ni] = __builtin_amdgcn_wmma_f32_16x16x32_bf16(
                    false, fa[mi].v, false, fb[ni].v,
                    (short)0, acc[mi][ni], false, false);
    };

    const int nkt = K >> 5;

    // prologue: stage K-tile 0 into buffer 0
    if constexpr (ASYNC) {
        stageAsync(0, 0);
        asm volatile("s_wait_asynccnt 0x0" ::: "memory");
    } else {
        loadRegs(0);
        storeRegs(0);
    }
    __syncthreads();

    for (int kt = 0; kt < nkt; ++kt) {
        const bool more = (kt + 1 < nkt);
        if constexpr (ASYNC) {
            if (more) stageAsync((kt + 1) & 1, (kt + 1) << 5);   // fire & forget
            compute(kt & 1);                                     // overlaps DMA
            asm volatile("s_wait_asynccnt 0x0" ::: "memory");
        } else {
            if (more) loadRegs((kt + 1) << 5);                   // loads in flight
            compute(kt & 1);                                     // overlaps loads
            if (more) storeRegs((kt + 1) & 1);
        }
        __syncthreads();
    }

    // epilogue: D layout = VGPR r -> M r (lanes 0-15) / r+8 (lanes 16-31)
    const int mAdd = (lane >> 4) << 3;
    #pragma unroll
    for (int mi = 0; mi < 4; ++mi) {
        #pragma unroll
        for (int ni = 0; ni < 2; ++ni) {
            const int gn = blockIdx.x * 128 + waveN * 32 + ni * 16 + (lane & 15);
            const float bv = bias[gn];
            #pragma unroll
            for (int r = 0; r < 8; ++r) {
                const long gm = (long)blockIdx.y * 128 + waveM * 64 + mi * 16 + r + mAdd;
                if (FULL || gm < M) {
                    const float val = acc[mi][ni][r] + bv;
                    if constexpr (OUT_BF16)
                        ((__bf16*)outv)[gm * (long)N + gn] = (__bf16)val;
                    else
                        ((float*)outv)[gm * (long)N + gn] = val;
                }
            }
        }
    }
}

// fp32 -> bf16 elementwise (weights)
__global__ void f2bf_kernel(const float* __restrict__ in, __bf16* __restrict__ out, long n)
{
    long i = (long)blockIdx.x * blockDim.x + threadIdx.x;
    long stride = (long)gridDim.x * blockDim.x;
    for (; i < n; i += stride) out[i] = (__bf16)in[i];
}

// adaptive_avg_pool1d over N (torch-exact segment bounds), result (B*9, C) bf16
__global__ void pool_kernel(const float* __restrict__ x, __bf16* __restrict__ pooled,
                            int NT, int C)
{
    const int c  = blockIdx.x * blockDim.x + threadIdx.x;   // channel
    const int bi = blockIdx.y;                              // b*9 + i
    const int i  = bi % 9;
    const int b  = bi / 9;
    const int s  = (i * NT) / 9;
    const int e  = ((i + 1) * NT + 8) / 9;
    float acc = 0.f;
    const float* p = x + ((long)b * NT) * C + c;
    for (int n = s; n < e; ++n) acc += p[(long)n * C];
    pooled[(long)bi * C + c] = (__bf16)(acc / (float)(e - s));
}

// SiLU + 9x9 kernel-gen + residual-mean subtraction, one thread per (b,c)
__global__ void kergen_kernel(const float* __restrict__ g, const float* __restrict__ kg_w,
                              const float* __restrict__ kg_b, const float* __restrict__ beta,
                              float* __restrict__ kern, int B, int C)
{
    const long idx = (long)blockIdx.x * blockDim.x + threadIdx.x;
    if (idx >= (long)B * C) return;
    const int b = (int)(idx / C);
    const int c = (int)(idx % C);

    float t[9];
    #pragma unroll
    for (int i = 0; i < 9; ++i) {
        const float v = g[((long)b * 9 + i) * C + c];
        t[i] = v / (1.f + __expf(-v));                      // SiLU
    }
    float ker[9];
    float mean = 0.f;
    #pragma unroll
    for (int j = 0; j < 9; ++j) {
        float a = kg_b[j];
        #pragma unroll
        for (int i = 0; i < 9; ++i) a += t[i] * kg_w[j * 9 + i];
        ker[j] = a;
        mean += a;
    }
    mean *= (1.f / 9.f);
    const float factor = 1.f / (1.f + __expf(-beta[c]));    // sigmoid(beta)
    #pragma unroll
    for (int j = 0; j < 9; ++j) kern[idx * 9 + j] = ker[j] - factor * mean;
}

// depthwise 3x3 conv, per-(b,c) kernels; x1/y layout (B, N, C) channel-contiguous
__global__ void dwconv_kernel(const __bf16* __restrict__ x1, const float* __restrict__ kern,
                              __bf16* __restrict__ y, int C)
{
    const int c = blockIdx.x * blockDim.x + threadIdx.x;
    const int n = blockIdx.y;                               // 0..1023
    const int b = blockIdx.z;                               // 0..31
    const int h = n >> 5, w = n & 31;

    float kv[9];
    const float* kp = kern + ((long)b * C + c) * 9;
    #pragma unroll
    for (int j = 0; j < 9; ++j) kv[j] = kp[j];

    float acc = 0.f;
    #pragma unroll
    for (int dy = -1; dy <= 1; ++dy) {
        const int hh = h + dy;
        if ((unsigned)hh >= 32u) continue;
        #pragma unroll
        for (int dx = -1; dx <= 1; ++dx) {
            const int ww = w + dx;
            if ((unsigned)ww >= 32u) continue;
            const long nn = (long)b * 1024 + (hh << 5) + ww;
            acc += (float)x1[nn * C + c] * kv[(dy + 1) * 3 + (dx + 1)];
        }
    }
    y[((long)b * 1024 + n) * C + c] = (__bf16)acc;
}

extern "C" void kernel_launch(void* const* d_in, const int* in_sizes, int n_in,
                              void* d_out, int out_size, void* d_ws, size_t ws_size,
                              hipStream_t stream)
{
    (void)in_sizes; (void)n_in; (void)out_size; (void)ws_size;
    constexpr int B = 32, NT = 1024, C = 768;
    constexpr long CC = (long)C * C;
    constexpr long BNC = (long)B * NT * C;

    const float* x      = (const float*)d_in[0];
    const float* p1_w   = (const float*)d_in[1];
    const float* p1_b   = (const float*)d_in[2];
    const float* kg_w   = (const float*)d_in[3];
    const float* kg_b   = (const float*)d_in[4];
    const float* p2_w   = (const float*)d_in[5];
    const float* p2_b   = (const float*)d_in[6];
    const float* proj_w = (const float*)d_in[7];
    const float* proj_b = (const float*)d_in[8];
    const float* beta   = (const float*)d_in[9];
    float* out = (float*)d_out;

    // workspace carve (256B aligned):
    char* wsp = (char*)d_ws;
    auto carve = [&](size_t bytes) -> char* {
        char* p = wsp;
        wsp += (bytes + 255) & ~(size_t)255;
        return p;
    };
    __bf16* w1bf   = (__bf16*)carve(CC * sizeof(__bf16));      // p1_w  bf16
    __bf16* w2bf   = (__bf16*)carve(CC * sizeof(__bf16));      // p2_w  bf16
    __bf16* wpbf   = (__bf16*)carve(CC * sizeof(__bf16));      // proj_w bf16
    __bf16* pooled = (__bf16*)carve((long)B * 9 * C * sizeof(__bf16));
    float*  g      = (float*)carve((long)B * 9 * C * sizeof(float));
    float*  kern   = (float*)carve((long)B * C * 9 * sizeof(float));
    __bf16* x1bf   = (__bf16*)carve(BNC * sizeof(__bf16));
    __bf16* ybf    = (__bf16*)carve(BNC * sizeof(__bf16));

    // 0) weights fp32 -> bf16
    {
        const int grid = (int)((CC + 255) / 256);
        f2bf_kernel<<<grid, 256, 0, stream>>>(p1_w,   w1bf, CC);
        f2bf_kernel<<<grid, 256, 0, stream>>>(p2_w,   w2bf, CC);
        f2bf_kernel<<<grid, 256, 0, stream>>>(proj_w, wpbf, CC);
    }

    // 1) adaptive pool of x over N: (B*9, C) bf16
    pool_kernel<<<dim3(C / 256, B * 9), 256, 0, stream>>>(x, pooled, NT, C);

    // 2) tiny GEMM: g = pooled * p1_w^T + p1_b  (M=288, fp32 out, guarded)
    gemm_wmma_bf16<false, false, false>
        <<<dim3(C / 128, (B * 9 + 127) / 128), 256, 0, stream>>>(
        pooled, w1bf, p1_b, g, B * 9, C, C);

    // 3) big GEMM: x1 = x * p2_w^T + p2_b  (bf16 out, fp32 A cvt-on-load, full)
    gemm_wmma_bf16<true, true, true>
        <<<dim3(C / 128, (B * NT) / 128), 256, 0, stream>>>(
        x, w2bf, p2_b, x1bf, B * NT, C, C);

    // 4) kernel generation: SiLU + 9x9 matmul + mean subtraction
    kergen_kernel<<<(B * C + 255) / 256, 256, 0, stream>>>(g, kg_w, kg_b, beta, kern, B, C);

    // 5) depthwise 3x3 conv
    dwconv_kernel<<<dim3(C / 256, NT, B), 256, 0, stream>>>(x1bf, kern, ybf, C);

    // 6) big GEMM: out = y * proj_w^T + proj_b  (fp32 out, bf16 A, ASYNC staging)
    gemm_wmma_bf16<false, false, true>
        <<<dim3(C / 128, (B * NT) / 128), 256, 0, stream>>>(
        ybf, wpbf, proj_b, out, B * NT, C, C);
}
